// LabelDependent_39694087750083
// MI455X (gfx1250) — matrix operand order repaired and checked by reading, not compiled, not run on hardware
//
#include <hip/hip_runtime.h>
#include <hip/hip_bf16.h>

typedef _Float16 half_t;
typedef __attribute__((ext_vector_type(16))) _Float16 v16h;
typedef __attribute__((ext_vector_type(8)))  float    v8f;

static constexpr int NB = 16;      // n_blocks
static constexpr int S  = 1024;    // seqlen
static constexpr int H  = 768;
static constexpr int C  = 64;
static constexpr int L  = 64;
static constexpr int M  = 4096;
static constexpr int U  = 2048;
static constexpr int T  = NB * S;  // 16384

union V8 { v8f v; float f[8]; };
union A16 { v16h v; unsigned int u[8]; };

__device__ __forceinline__ float waveReduceSum(float v) {
  #pragma unroll
  for (int off = 16; off > 0; off >>= 1) v += __shfl_down(v, off, 32);
  return v;
}

// ---------- flatT[h*T + t] = (f16) flat[t*H + h] ----------
__global__ void transpose_kernel(const float* __restrict__ flat, half_t* __restrict__ flatT) {
  size_t i = (size_t)blockIdx.x * blockDim.x + threadIdx.x;
  if (i >= (size_t)T * H) return;
  int t = (int)(i % T);
  int h = (int)(i / T);
  flatT[(size_t)h * T + t] = (half_t)flat[(size_t)t * H + h];
}

// ---------- out[r] = x[r,:]·w + bias  (wave per row) ----------
__global__ void dot_vec_kernel(const float* __restrict__ x, const float* __restrict__ w,
                               const float* __restrict__ bias, float* __restrict__ out, int nrows) {
  int wv = threadIdx.x >> 5, lane = threadIdx.x & 31;
  int r = blockIdx.x * 8 + wv;
  if (r >= nrows) return;
  const float* xp = x + (size_t)r * H;
  float acc = 0.f;
  for (int h = lane; h < H; h += 32) acc += xp[h] * w[h];
  acc = waveReduceSum(acc);
  if (lane == 0) out[r] = acc + bias[0];
}

// ---------- uscore[l*T+t] = labels[l]·flat[t]  (wave per pair) ----------
__global__ void uscore_kernel(const float* __restrict__ labels, const float* __restrict__ flat,
                              float* __restrict__ uscore) {
  int wv = threadIdx.x >> 5, lane = threadIdx.x & 31;
  int p = blockIdx.x * 8 + wv;              // [0, L*T)
  int l = p >> 14;                          // T = 2^14
  int t = p & (T - 1);
  const float* lp = labels + (size_t)l * H;
  const float* fp = flat + (size_t)t * H;
  float acc = 0.f;
  for (int h = lane; h < H; h += 32) acc += lp[h] * fp[h];
  acc = waveReduceSum(acc);
  if (lane == 0) uscore[(size_t)l * T + t] = acc;
}

// ---------- cmask[c,t] = 1 for t in any utterance range of c ----------
__global__ void cmask_kernel(const int* __restrict__ uidx, const int* __restrict__ ucid,
                             unsigned char* __restrict__ cmask) {
  int u = blockIdx.x;
  int a = uidx[2 * u], b = uidx[2 * u + 1];
  int c = ucid[u];
  for (int t = a + threadIdx.x; t < b; t += blockDim.x) cmask[(size_t)c * T + t] = 1;
}

// ---------- mention softmax stats (range-closed-form) ----------
__global__ void mstats_kernel(const float* __restrict__ ms, const int* __restrict__ midx,
                              float* __restrict__ mmx, float* __restrict__ minvZ) {
  __shared__ float red[256];
  int m = blockIdx.x, tid = threadIdx.x;
  int a = midx[2 * m], b = midx[2 * m + 1];
  int len = b - a;
  float mx = -1e30f;
  for (int t = a + tid; t < b; t += 256) mx = fmaxf(mx, ms[t]);
  red[tid] = mx; __syncthreads();
  for (int s = 128; s > 0; s >>= 1) { if (tid < s) red[tid] = fmaxf(red[tid], red[tid + s]); __syncthreads(); }
  mx = red[0];
  if (len < T) mx = fmaxf(mx, 0.f);       // masked-out entries contribute value 0
  __syncthreads();
  float sum = 0.f;
  for (int t = a + tid; t < b; t += 256) sum += __expf(ms[t] - mx);
  red[tid] = sum; __syncthreads();
  for (int s = 128; s > 0; s >>= 1) { if (tid < s) red[tid] += red[tid + s]; __syncthreads(); }
  if (tid == 0) {
    float Z = red[0] + (float)(T - len) * __expf(-mx);
    mmx[m] = mx; minvZ[m] = 1.f / Z;
  }
}

// ---------- utterance softmax stats per (c,l) row ----------
__global__ void ustats_kernel(const float* __restrict__ uscore, const unsigned char* __restrict__ cmask,
                              float* __restrict__ umx, float* __restrict__ uinvZ) {
  __shared__ float red[256];
  int r = blockIdx.x, tid = threadIdx.x;
  int c = r >> 6, l = r & 63;
  float mx = -1e30f;
  for (int t = tid; t < T; t += 256) {
    float v = cmask[(size_t)c * T + t] ? uscore[(size_t)l * T + t] : 0.f;
    mx = fmaxf(mx, v);
  }
  red[tid] = mx; __syncthreads();
  for (int s = 128; s > 0; s >>= 1) { if (tid < s) red[tid] = fmaxf(red[tid], red[tid + s]); __syncthreads(); }
  mx = red[0]; __syncthreads();
  float sum = 0.f;
  for (int t = tid; t < T; t += 256) {
    float v = cmask[(size_t)c * T + t] ? uscore[(size_t)l * T + t] : 0.f;
    sum += __expf(v - mx);
  }
  red[tid] = sum; __syncthreads();
  for (int s = 128; s > 0; s >>= 1) { if (tid < s) red[tid] += red[tid + s]; __syncthreads(); }
  if (tid == 0) { umx[r] = mx; uinvZ[r] = 1.f / red[0]; }
}

// ---------- fused WMMA weighted sum: m_emb[M,H] = softmax(mask·s)[M,T] @ flat[T,H] ----------
// K step 64, double-buffered A tile (one barrier per 64-K), 3 H tiles per wave.
__global__ __launch_bounds__(512)
void wsum_mention_kernel(const float* __restrict__ mscore, const int* __restrict__ midx,
                         const float* __restrict__ mmx, const float* __restrict__ minvZ,
                         const half_t* __restrict__ flatT, float* __restrict__ m_emb) {
  __shared__ half_t aTile[2][16 * 64];
  __shared__ int sA[16], sB[16];
  __shared__ float sMx[16], sIZ[16];
  const int tid = threadIdx.x;
  const int mtile = blockIdx.x;
  if (tid < 16) {
    int m = mtile * 16 + tid;
    sA[tid] = midx[2 * m]; sB[tid] = midx[2 * m + 1];
    sMx[tid] = mmx[m];     sIZ[tid] = minvZ[m];
  }
  __syncthreads();
  const int lane = tid & 31;
  const int wv   = tid >> 5;     // wave id 0..15
  const int n    = lane & 15;
  const int hs   = lane >> 4;
  V8 acc[3] = {};
  int p = 0;
  for (int kt = 0; kt < T; kt += 64) {
    half_t* buf = &aTile[p][0];
    // fill 16x64 weight tile (2 elements per thread)
    {
      int e = tid;
      int rw = e >> 6, k = e & 63, t = kt + k;
      float v = (sA[rw] <= t && t < sB[rw]) ? mscore[t] : 0.f;
      buf[rw * 64 + k] = (half_t)__expf(v - sMx[rw]);
      e = tid + 512; rw = e >> 6; k = e & 63; t = kt + k;
      v = (sA[rw] <= t && t < sB[rw]) ? mscore[t] : 0.f;
      buf[rw * 64 + k] = (half_t)__expf(v - sMx[rw]);
    }
    __syncthreads();
    #pragma unroll
    for (int kh = 0; kh < 2; kh++) {
      A16 af;
      #pragma unroll
      for (int i = 0; i < 4; i++)
        af.u[i] = *(const unsigned int*)&buf[(lane & 15) * 64 + kh * 32 + hs * 16 + 2 * i];
      #pragma unroll
      for (int i = 0; i < 4; i++)
        af.u[4 + i] = *(const unsigned int*)&buf[(lane & 15) * 64 + kh * 32 + hs * 16 + 8 + 2 * i];
      #pragma unroll
      for (int j = 0; j < 3; j++) {
        int h0 = (wv + 16 * j) * 16;
        const half_t* bp = flatT + (size_t)(h0 + n) * T + kt + kh * 32 + hs * 16;
        v16h bf = *(const v16h*)bp;
        acc[j].v = __builtin_amdgcn_wmma_f32_16x16x32_f16(false, af.v, false, bf,
                                                          (short)0, acc[j].v, false, false);
      }
    }
    p ^= 1;
  }
  #pragma unroll
  for (int j = 0; j < 3; j++) {
    int h0 = (wv + 16 * j) * 16;
    #pragma unroll
    for (int e = 0; e < 8; e++) {
      int ml = hs * 8 + e;
      m_emb[(size_t)(mtile * 16 + ml) * H + h0 + n] = acc[j].f[e] * sIZ[ml];
    }
  }
}

// ---------- fused WMMA weighted sum: utt_emb[C*L,H] = softmax(mask·u)[C*L,T] @ flat[T,H] ----------
__global__ __launch_bounds__(512)
void wsum_utter_kernel(const float* __restrict__ uscore, const unsigned char* __restrict__ cmask,
                       const float* __restrict__ umx, const float* __restrict__ uinvZ,
                       const half_t* __restrict__ flatT, float* __restrict__ utt_emb) {
  __shared__ half_t aTile[2][16 * 64];
  __shared__ float sMx[16], sIZ[16];
  const int tid = threadIdx.x;
  const int rtile = blockIdx.x;
  const int c  = rtile >> 2;
  const int l0 = (rtile & 3) * 16;
  if (tid < 16) {
    int r = rtile * 16 + tid;
    sMx[tid] = umx[r]; sIZ[tid] = uinvZ[r];
  }
  __syncthreads();
  const int lane = tid & 31;
  const int wv   = tid >> 5;
  const int n    = lane & 15;
  const int hs   = lane >> 4;
  V8 acc[3] = {};
  int p = 0;
  for (int kt = 0; kt < T; kt += 64) {
    half_t* buf = &aTile[p][0];
    {
      int e = tid;
      int rw = e >> 6, k = e & 63, t = kt + k;
      float v = cmask[(size_t)c * T + t] ? uscore[(size_t)(l0 + rw) * T + t] : 0.f;
      buf[rw * 64 + k] = (half_t)__expf(v - sMx[rw]);
      e = tid + 512; rw = e >> 6; k = e & 63; t = kt + k;
      v = cmask[(size_t)c * T + t] ? uscore[(size_t)(l0 + rw) * T + t] : 0.f;
      buf[rw * 64 + k] = (half_t)__expf(v - sMx[rw]);
    }
    __syncthreads();
    #pragma unroll
    for (int kh = 0; kh < 2; kh++) {
      A16 af;
      #pragma unroll
      for (int i = 0; i < 4; i++)
        af.u[i] = *(const unsigned int*)&buf[(lane & 15) * 64 + kh * 32 + hs * 16 + 2 * i];
      #pragma unroll
      for (int i = 0; i < 4; i++)
        af.u[4 + i] = *(const unsigned int*)&buf[(lane & 15) * 64 + kh * 32 + hs * 16 + 8 + 2 * i];
      #pragma unroll
      for (int j = 0; j < 3; j++) {
        int h0 = (wv + 16 * j) * 16;
        const half_t* bp = flatT + (size_t)(h0 + n) * T + kt + kh * 32 + hs * 16;
        v16h bf = *(const v16h*)bp;
        acc[j].v = __builtin_amdgcn_wmma_f32_16x16x32_f16(false, af.v, false, bf,
                                                          (short)0, acc[j].v, false, false);
      }
    }
    p ^= 1;
  }
  #pragma unroll
  for (int j = 0; j < 3; j++) {
    int h0 = (wv + 16 * j) * 16;
    #pragma unroll
    for (int e = 0; e < 8; e++) {
      int ml = hs * 8 + e;
      utt_emb[(size_t)(rtile * 16 + ml) * H + h0 + n] = acc[j].f[e] * sIZ[ml];
    }
  }
}

// ---------- names attention per (c,b): softmax over S, weighted sum of story ----------
__global__ void names_kernel(const float* __restrict__ nscore, const int* __restrict__ nidx,
                             const float* __restrict__ story, float* __restrict__ names_emb) {
  __shared__ float red[256];
  __shared__ float wbuf[S];
  int cb = blockIdx.x;
  int c = cb >> 4, b = cb & 15;
  int tid = threadIdx.x;
  int a = nidx[2 * c], e = nidx[2 * c + 1];
  float mx = -1e30f;
  for (int s2 = tid; s2 < S; s2 += 256) {
    float sc = nscore[b * S + s2];
    float v = (a <= s2 && s2 < e) ? sc : 0.f;
    mx = fmaxf(mx, v);
  }
  red[tid] = mx; __syncthreads();
  for (int s = 128; s > 0; s >>= 1) { if (tid < s) red[tid] = fmaxf(red[tid], red[tid + s]); __syncthreads(); }
  mx = red[0]; __syncthreads();
  float sum = 0.f;
  for (int s2 = tid; s2 < S; s2 += 256) {
    float sc = nscore[b * S + s2];
    float v = (a <= s2 && s2 < e) ? sc : 0.f;
    float w = __expf(v - mx);
    wbuf[s2] = w; sum += w;
  }
  red[tid] = sum; __syncthreads();
  for (int s = 128; s > 0; s >>= 1) { if (tid < s) red[tid] += red[tid + s]; __syncthreads(); }
  float iZ = 1.f / red[0];
  for (int h = tid; h < H; h += 256) {
    float acc = 0.f;
    for (int s2 = 0; s2 < S; s2++) acc += wbuf[s2] * story[((size_t)b * S + s2) * H + h];
    names_emb[((size_t)c * NB + b) * H + h] = acc * iZ;
  }
}

// ---------- name-label attention per (c,l): softmax over B=16 ----------
__global__ void nl_kernel(const float* __restrict__ names_emb, const float* __restrict__ labels,
                          float* __restrict__ nl_emb) {
  __shared__ float sc[16];
  int r = blockIdx.x;
  int c = r >> 6, l = r & 63;
  int tid = threadIdx.x, wv = tid >> 5, lane = tid & 31;
  for (int b = wv; b < 16; b += 8) {
    const float* ep = names_emb + ((size_t)c * NB + b) * H;
    const float* lp = labels + (size_t)l * H;
    float acc = 0.f;
    for (int h = lane; h < H; h += 32) acc += ep[h] * lp[h];
    acc = waveReduceSum(acc);
    if (lane == 0) sc[b] = acc;
  }
  __syncthreads();
  float mx = -1e30f;
  #pragma unroll
  for (int b = 0; b < 16; b++) mx = fmaxf(mx, sc[b]);
  float w[16]; float Z = 0.f;
  #pragma unroll
  for (int b = 0; b < 16; b++) { w[b] = __expf(sc[b] - mx); Z += w[b]; }
  float iZ = 1.f / Z;
  for (int h = tid; h < H; h += 256) {
    float acc = 0.f;
    #pragma unroll
    for (int b = 0; b < 16; b++) acc += w[b] * names_emb[((size_t)c * NB + b) * H + h];
    nl_emb[(size_t)r * H + h] = acc * iZ;
  }
}

// ---------- ml_score[m,l] = m_emb[m]·labels[l] ----------
__global__ void mlscore_kernel(const float* __restrict__ m_emb, const float* __restrict__ labels,
                               float* __restrict__ mls) {
  int wv = threadIdx.x >> 5, lane = threadIdx.x & 31;
  int p = blockIdx.x * 8 + wv;   // [0, M*L)
  int m = p >> 6, l = p & 63;
  const float* ep = m_emb + (size_t)m * H;
  const float* lp = labels + (size_t)l * H;
  float acc = 0.f;
  for (int h = lane; h < H; h += 32) acc += ep[h] * lp[h];
  acc = waveReduceSum(acc);
  if (lane == 0) mls[(size_t)m * L + l] = acc;
}

// ---------- mention-label attention per (c,l): masked softmax over that character's mentions ----------
__global__ void mention_label_kernel(const float* __restrict__ mls, const int* __restrict__ mcid,
                                     const float* __restrict__ m_emb, float* __restrict__ men_emb,
                                     int* __restrict__ hasM) {
  __shared__ float red[256];
  int r = blockIdx.x, tid = threadIdx.x;
  int c = r >> 6, l = r & 63;
  float mx = -1e30f; int cnt = 0;
  for (int m = tid; m < M; m += 256)
    if (mcid[m] == c) { mx = fmaxf(mx, mls[(size_t)m * L + l]); cnt++; }
  red[tid] = mx; __syncthreads();
  for (int s = 128; s > 0; s >>= 1) { if (tid < s) red[tid] = fmaxf(red[tid], red[tid + s]); __syncthreads(); }
  mx = red[0]; __syncthreads();
  red[tid] = (float)cnt; __syncthreads();
  for (int s = 128; s > 0; s >>= 1) { if (tid < s) red[tid] += red[tid + s]; __syncthreads(); }
  int total = (int)red[0]; __syncthreads();
  if (total == 0) {
    for (int h = tid; h < H; h += 256) men_emb[(size_t)r * H + h] = 0.f;
    if (tid == 0 && l == 0) hasM[c] = 0;
    return;
  }
  float sum = 0.f;
  for (int m = tid; m < M; m += 256)
    if (mcid[m] == c) sum += __expf(mls[(size_t)m * L + l] - mx);
  red[tid] = sum; __syncthreads();
  for (int s = 128; s > 0; s >>= 1) { if (tid < s) red[tid] += red[tid + s]; __syncthreads(); }
  float iZ = 1.f / red[0];
  float a0 = 0.f, a1 = 0.f, a2 = 0.f;
  for (int m = 0; m < M; m++) {
    if (mcid[m] == c) {
      float w = __expf(mls[(size_t)m * L + l] - mx) * iZ;
      const float* ep = m_emb + (size_t)m * H;
      a0 += w * ep[tid]; a1 += w * ep[tid + 256]; a2 += w * ep[tid + 512];
    }
  }
  men_emb[(size_t)r * H + tid]       = a0;
  men_emb[(size_t)r * H + tid + 256] = a1;
  men_emb[(size_t)r * H + tid + 512] = a2;
  if (tid == 0 && l == 0) hasM[c] = 1;
}

// ---------- has_utt[c] ----------
__global__ void hasutt_kernel(const unsigned char* __restrict__ cmask, int* __restrict__ hasU) {
  __shared__ int any;
  int c = blockIdx.x, tid = threadIdx.x;
  if (tid == 0) any = 0;
  __syncthreads();
  int loc = 0;
  for (int t = tid; t < T; t += 256) loc |= cmask[(size_t)c * T + t];
  if (loc) atomicOr(&any, 1);
  __syncthreads();
  if (tid == 0) hasU[c] = any;
}

// ---------- combine per (c,l) ----------
__global__ void combine_kernel(const float* __restrict__ nl_emb, const float* __restrict__ men_emb,
                               const float* __restrict__ utt_emb, const float* __restrict__ wc,
                               const float* __restrict__ bc, const int* __restrict__ hasM,
                               const int* __restrict__ hasU, float* __restrict__ out) {
  __shared__ float sc[3];
  int r = blockIdx.x;
  int c = r >> 6;
  int tid = threadIdx.x, wv = tid >> 5, lane = tid & 31;
  const float* e0 = nl_emb  + (size_t)r * H;
  const float* e1 = men_emb + (size_t)r * H;
  const float* e2 = utt_emb + (size_t)r * H;
  if (wv < 3) {
    const float* ep = (wv == 0) ? e0 : (wv == 1 ? e1 : e2);
    float acc = 0.f;
    for (int h = lane; h < H; h += 32) acc += ep[h] * wc[h];
    acc = waveReduceSum(acc);
    if (lane == 0) sc[wv] = acc + bc[0];
  }
  __syncthreads();
  float m1 = hasM[c] ? 1.f : 0.f;
  float m2 = hasU[c] ? 1.f : 0.f;
  float v0 = sc[0], v1 = sc[1] * m1, v2 = sc[2] * m2;
  float mx = fmaxf(v0, fmaxf(v1, v2));
  float w0 = __expf(v0 - mx), w1 = __expf(v1 - mx), w2 = __expf(v2 - mx);
  float iZ = 1.f / (w0 + w1 + w2);
  for (int h = tid; h < H; h += 256)
    out[(size_t)r * H + h] = (w0 * e0[h] + w1 * e1[h] + w2 * e2[h]) * iZ;
}

static inline char* carve(char*& p, size_t bytes) {
  char* r = p;
  p += (bytes + 255) & ~(size_t)255;
  return r;
}

extern "C" void kernel_launch(void* const* d_in, const int* in_sizes, int n_in,
                              void* d_out, int out_size, void* d_ws, size_t ws_size,
                              hipStream_t stream) {
  (void)in_sizes; (void)n_in; (void)out_size; (void)ws_size;
  const float* story     = (const float*)d_in[0];     // [B,S,H] == flat [T,H]
  const int*   names_idx = (const int*)d_in[1];
  const int*   ments_idx = (const int*)d_in[2];
  const int*   utts_idx  = (const int*)d_in[3];
  const int*   mcid      = (const int*)d_in[4];
  const int*   ucid      = (const int*)d_in[5];
  const float* labels    = (const float*)d_in[6];
  const float* w_name    = (const float*)d_in[7];
  const float* b_name    = (const float*)d_in[8];
  const float* w_mention = (const float*)d_in[9];
  const float* b_mention = (const float*)d_in[10];
  const float* w_combine = (const float*)d_in[11];
  const float* b_combine = (const float*)d_in[12];
  float* out = (float*)d_out;

  char* p = (char*)d_ws;
  half_t*        flatT   = (half_t*)       carve(p, (size_t)T * H * sizeof(half_t));
  float*         mscore  = (float*)        carve(p, (size_t)T * sizeof(float));
  float*         nscore  = (float*)        carve(p, (size_t)T * sizeof(float));
  float*         uscore  = (float*)        carve(p, (size_t)L * T * sizeof(float));
  unsigned char* cmask   = (unsigned char*)carve(p, (size_t)C * T);
  float*         mmx     = (float*)        carve(p, (size_t)M * sizeof(float));
  float*         minvZ   = (float*)        carve(p, (size_t)M * sizeof(float));
  float*         umx     = (float*)        carve(p, (size_t)C * L * sizeof(float));
  float*         uinvZ   = (float*)        carve(p, (size_t)C * L * sizeof(float));
  float*         m_emb   = (float*)        carve(p, (size_t)M * H * sizeof(float));
  float*         utt_emb = (float*)        carve(p, (size_t)C * L * H * sizeof(float));
  float*         nm_emb  = (float*)        carve(p, (size_t)C * NB * H * sizeof(float));
  float*         nl_emb  = (float*)        carve(p, (size_t)C * L * H * sizeof(float));
  float*         men_emb = (float*)        carve(p, (size_t)C * L * H * sizeof(float));
  float*         mls     = (float*)        carve(p, (size_t)M * L * sizeof(float));
  int*           hasM    = (int*)          carve(p, (size_t)C * sizeof(int));
  int*           hasU    = (int*)          carve(p, (size_t)C * sizeof(int));

  transpose_kernel<<<(T * H + 255) / 256, 256, 0, stream>>>(story, flatT);
  dot_vec_kernel<<<T / 8, 256, 0, stream>>>(story, w_mention, b_mention, mscore, T);
  dot_vec_kernel<<<T / 8, 256, 0, stream>>>(story, w_name, b_name, nscore, T);
  uscore_kernel<<<(L * T) / 8, 256, 0, stream>>>(labels, story, uscore);
  hipMemsetAsync(cmask, 0, (size_t)C * T, stream);
  cmask_kernel<<<U, 256, 0, stream>>>(utts_idx, ucid, cmask);
  mstats_kernel<<<M, 256, 0, stream>>>(mscore, ments_idx, mmx, minvZ);
  ustats_kernel<<<C * L, 256, 0, stream>>>(uscore, cmask, umx, uinvZ);
  wsum_mention_kernel<<<M / 16, 512, 0, stream>>>(mscore, ments_idx, mmx, minvZ, flatT, m_emb);
  wsum_utter_kernel<<<(C * L) / 16, 512, 0, stream>>>(uscore, cmask, umx, uinvZ, flatT, utt_emb);
  names_kernel<<<C * NB, 256, 0, stream>>>(nscore, names_idx, story, nm_emb);
  nl_kernel<<<C * L, 256, 0, stream>>>(nm_emb, labels, nl_emb);
  mlscore_kernel<<<(M * L) / 8, 256, 0, stream>>>(m_emb, labels, mls);
  mention_label_kernel<<<C * L, 256, 0, stream>>>(mls, mcid, m_emb, men_emb, hasM);
  hasutt_kernel<<<C, 256, 0, stream>>>(cmask, hasU);
  combine_kernel<<<C * L, 256, 0, stream>>>(nl_emb, men_emb, utt_emb, w_combine, b_combine,
                                            hasM, hasU, out);
}
